// ThousandColumns_14542759264917
// MI455X (gfx1250) — matrix operand order repaired and checked by reading, not compiled, not run on hardware
//
#include <hip/hip_runtime.h>
#include <math.h>

// ---------------- model dims ----------------
#define BB   2
#define TT   4
#define NN   2048
#define DSZ  64      // d_state
#define DIZ  256     // d_input
#define DVZ  64      // d_vote
#define KACT 512     // top-k active
#define NH   4       // heads
#define HDZ  16      // head dim

typedef __attribute__((ext_vector_type(16))) _Float16 v16h;
typedef __attribute__((ext_vector_type(8)))  float    v8f;

// ---------------- WMMA helpers (gfx1250, wave32) ----------------
__device__ __forceinline__ v8f wmma16(v16h a, v16h b, v8f c) {
  // D(16x16,f32) = A(16x32,f16) x B(32x16,f16) + C
  return __builtin_amdgcn_wmma_f32_16x16x32_f16(false, a, false, b, (short)0, c,
                                                false, false);
}

// A fragment from an f32 tile in LDS (16 rows x row_stride cols), K-offset = ktile*32.
// Layout (this code's contiguous k order, not the ISA table order; 16-bit A 16x32): lane L: M=L&15, half=L>>4;
// VGPR j holds K = 16*half + 2j  (pair K, K+1).
__device__ __forceinline__ v16h load_afrag(const float* tile, int row_stride,
                                           int ktile, int lane) {
  int M = lane & 15;
  int half = lane >> 4;
  const float* row = tile + M * row_stride + ktile * 32;
  v16h a;
#pragma unroll
  for (int j = 0; j < 8; ++j) {
    int kb = half * 16 + 2 * j;
    a[2 * j]     = (_Float16)row[kb];
    a[2 * j + 1] = (_Float16)row[kb + 1];
  }
  return a;
}

// B fragment: preconverted in global memory, 32B contiguous per lane per tile.
__device__ __forceinline__ v16h load_bfrag(const _Float16* wbuf, int tile, int lane) {
  return *(const v16h*)(wbuf + ((size_t)tile * 32 + (size_t)lane) * 16);
}

// ---------------- weight prep: f32 W (rows=outN, cols=inK) -> B-frag f16 ----------------
// B[k][n] = W[row_off+n][k].  tile = ntile*numK + ktile; lane L: n = ntile*16 + (L&15),
// half = L>>4; slot j holds K = ktile*32 + half*16 + 2j (+1).
__global__ void k_prep(const float* __restrict__ W, _Float16* __restrict__ dst,
                       int numK, int ld, int row_off) {
  int tile  = blockIdx.x;
  int ntile = tile / numK;
  int ktile = tile % numK;
  int lane  = threadIdx.x;
  int n     = ntile * 16 + (lane & 15);
  int half  = lane >> 4;
  _Float16* p = dst + ((size_t)tile * 32 + (size_t)lane) * 16;
#pragma unroll
  for (int j = 0; j < 8; ++j) {
    int k0 = ktile * 32 + half * 16 + 2 * j;
    p[2 * j]     = (_Float16)W[(size_t)(row_off + n) * ld + k0];
    p[2 * j + 1] = (_Float16)W[(size_t)(row_off + n) * ld + k0 + 1];
  }
}

// ---------------- gate + input projection ----------------
__global__ void k_gate(const float* __restrict__ x, const float* __restrict__ W_in,
                       const float* __restrict__ W_gate, const float* __restrict__ h,
                       float* __restrict__ gate, float* __restrict__ xproj, int t) {
  int i = blockIdx.x * blockDim.x + threadIdx.x;
  if (i < BB * NN) {
    int b = i / NN, n = i % NN;
    const float* hr = h + ((size_t)(b * NN + n)) * DSZ;
    float s = 0.f;
#pragma unroll 8
    for (int d = 0; d < DSZ; ++d) s += hr[d] * W_gate[d];
    gate[i] = s;
  } else if (i < BB * NN + BB * DSZ) {
    int j = i - BB * NN;
    int b = j / DSZ, d = j % DSZ;
    const float* xr = x + ((size_t)b * TT + t) * DIZ;
    float s = 0.f;
#pragma unroll 8
    for (int k = 0; k < DIZ; ++k) s += xr[k] * W_in[(size_t)d * DIZ + k];
    xproj[b * DSZ + d] = s;
  }
}

// ---------------- top-K hard mask (radix bisection on order-preserving keys) ----------------
__global__ void k_topk(const float* __restrict__ gate, float* __restrict__ mask) {
  int b   = blockIdx.x;
  int tid = threadIdx.x;
  __shared__ unsigned keys[NN];
  __shared__ int red[256];
  const float* g = gate + (size_t)b * NN;
  for (int i = tid; i < NN; i += 256) {
    unsigned u = __float_as_uint(g[i]);
    keys[i] = (u & 0x80000000u) ? ~u : (u | 0x80000000u);
  }
  __syncthreads();
  unsigned thr = 0u;
  for (int bit = 31; bit >= 0; --bit) {
    unsigned cand = thr | (1u << bit);
    int c = 0;
    for (int i = tid; i < NN; i += 256) c += (keys[i] >= cand) ? 1 : 0;
    red[tid] = c; __syncthreads();
    for (int s = 128; s > 0; s >>= 1) { if (tid < s) red[tid] += red[tid + s]; __syncthreads(); }
    int cnt = red[0]; __syncthreads();
    if (cnt >= KACT) thr = cand;
  }
  int cg = 0;
  for (int i = tid; i < NN; i += 256) cg += (keys[i] > thr) ? 1 : 0;
  red[tid] = cg; __syncthreads();
  for (int s = 128; s > 0; s >>= 1) { if (tid < s) red[tid] += red[tid + s]; __syncthreads(); }
  int G = red[0]; __syncthreads();
  int need = KACT - G;  // ties taken in ascending index order (jax top_k behavior)
  for (int i = tid; i < NN; i += 256) {
    float mv;
    if (keys[i] > thr) mv = 1.0f;
    else if (keys[i] == thr) {
      int rank = 0;
      for (int j = 0; j < i; ++j) rank += (keys[j] == thr) ? 1 : 0;
      mv = (rank < need) ? 1.0f : 0.0f;
    } else mv = 0.0f;
    mask[(size_t)b * NN + i] = mv;
  }
}

// ---------------- column core: rmsnorm -> SwiGLU (WMMA) -> h update -> vote+QKV (WMMA) ----
__global__ void __launch_bounds__(32)
k_core(float* __restrict__ h_g, const float* __restrict__ xproj,
       const float* __restrict__ mask_g, const float* __restrict__ norm_w,
       const _Float16* __restrict__ wg16, const _Float16* __restrict__ wout16,
       const _Float16* __restrict__ wvote16, const _Float16* __restrict__ wq16,
       const _Float16* __restrict__ wk16, const _Float16* __restrict__ wv16,
       float* __restrict__ v_g, float* __restrict__ q_g,
       float* __restrict__ k_g, float* __restrict__ vh_g) {
  __shared__ float zs[16 * 128];   // rmsnormed [h | xproj] (the zero m-slice is dropped)
  __shared__ float gs[16 * 256];   // a, then a*silu(b)
  __shared__ float hs[16 * 64];    // updated h tile
  __shared__ float vs[16 * 64];    // masked vote tile
  __shared__ float mrow[16];

  int lane = threadIdx.x;
  int b    = blockIdx.x / (NN / 16);
  int tile = blockIdx.x % (NN / 16);
  int row0 = tile * 16;

  const float* xp = xproj + (size_t)b * DSZ;
  float sxx = 0.f;
#pragma unroll 8
  for (int d = 0; d < DSZ; ++d) { float xv = xp[d]; sxx += xv * xv; }

  if (lane < 16) {
    mrow[lane] = mask_g[(size_t)b * NN + row0 + lane];
    const float* hr = h_g + ((size_t)(b * NN + row0 + lane)) * DSZ;
    float s = sxx;
#pragma unroll 8
    for (int d = 0; d < DSZ; ++d) { float hv = hr[d]; s += hv * hv; }
    float r = rsqrtf(s * (1.0f / 192.0f) + 1e-6f);  // mean over 192 (incl. zero m-slice)
#pragma unroll 8
    for (int d = 0; d < DSZ; ++d)  zs[lane * 128 + d]       = norm_w[d] * hr[d] * r;
#pragma unroll 8
    for (int d = 0; d < DSZ; ++d)  zs[lane * 128 + 64 + d]  = norm_w[64 + d] * xp[d] * r;
  }
  __syncthreads();

  int ncol = lane & 15, half = lane >> 4;

  // GEMM1: g = z @ core_Wg[:, :128]^T  (M=16, K=128, N=512)
  v16h af0 = load_afrag(zs, 128, 0, lane);
  v16h af1 = load_afrag(zs, 128, 1, lane);
  v16h af2 = load_afrag(zs, 128, 2, lane);
  v16h af3 = load_afrag(zs, 128, 3, lane);
  __builtin_prefetch(wout16 + (size_t)lane * 16, 0, 1);

  for (int ct = 0; ct < 16; ++ct) {      // 'a' half
    v8f acc = {};
    acc = wmma16(af0, load_bfrag(wg16, ct * 4 + 0, lane), acc);
    acc = wmma16(af1, load_bfrag(wg16, ct * 4 + 1, lane), acc);
    acc = wmma16(af2, load_bfrag(wg16, ct * 4 + 2, lane), acc);
    acc = wmma16(af3, load_bfrag(wg16, ct * 4 + 3, lane), acc);
#pragma unroll
    for (int r = 0; r < 8; ++r) gs[(half * 8 + r) * 256 + ct * 16 + ncol] = acc[r];
  }
  __syncthreads();
  for (int ct = 0; ct < 16; ++ct) {      // 'b' half -> hh = a*silu(b)
    v8f acc = {};
    acc = wmma16(af0, load_bfrag(wg16, (16 + ct) * 4 + 0, lane), acc);
    acc = wmma16(af1, load_bfrag(wg16, (16 + ct) * 4 + 1, lane), acc);
    acc = wmma16(af2, load_bfrag(wg16, (16 + ct) * 4 + 2, lane), acc);
    acc = wmma16(af3, load_bfrag(wg16, (16 + ct) * 4 + 3, lane), acc);
#pragma unroll
    for (int r = 0; r < 8; ++r) {
      int idx = (half * 8 + r) * 256 + ct * 16 + ncol;
      float av = gs[idx], bv = acc[r];
      gs[idx] = av * (bv / (1.0f + __expf(-bv)));
    }
  }
  __syncthreads();

  // GEMM2: delta = hh @ core_Wout^T (K=256, N=64); h += mask*delta
  v16h ag[8];
#pragma unroll
  for (int kt = 0; kt < 8; ++kt) ag[kt] = load_afrag(gs, 256, kt, lane);
#pragma unroll
  for (int ot = 0; ot < 4; ++ot) {
    v8f acc = {};
#pragma unroll
    for (int kt = 0; kt < 8; ++kt)
      acc = wmma16(ag[kt], load_bfrag(wout16, ot * 8 + kt, lane), acc);
#pragma unroll
    for (int r = 0; r < 8; ++r) {
      int M = half * 8 + r, col = ot * 16 + ncol;
      size_t gi = ((size_t)(b * NN + row0 + M)) * DSZ + col;
      float hn = h_g[gi] + mrow[M] * acc[r];
      hs[M * 64 + col] = hn;
      h_g[gi] = hn;
    }
  }
  __syncthreads();

  // GEMM3: v = (h_new @ W_vote^T) * mask
  v16h ah0 = load_afrag(hs, 64, 0, lane);
  v16h ah1 = load_afrag(hs, 64, 1, lane);
#pragma unroll
  for (int ot = 0; ot < 4; ++ot) {
    v8f acc = {};
    acc = wmma16(ah0, load_bfrag(wvote16, ot * 2 + 0, lane), acc);
    acc = wmma16(ah1, load_bfrag(wvote16, ot * 2 + 1, lane), acc);
#pragma unroll
    for (int r = 0; r < 8; ++r) {
      int M = half * 8 + r, col = ot * 16 + ncol;
      float vv = acc[r] * mrow[M];
      vs[M * 64 + col] = vv;
      v_g[((size_t)(b * NN + row0 + M)) * DVZ + col] = vv;
    }
  }
  __syncthreads();

  // QKV projections of v
  v16h av0 = load_afrag(vs, 64, 0, lane);
  v16h av1 = load_afrag(vs, 64, 1, lane);
  const _Float16* wts[3] = { wq16, wk16, wv16 };
  float* outs[3] = { q_g, k_g, vh_g };
#pragma unroll
  for (int p = 0; p < 3; ++p) {
#pragma unroll
    for (int ot = 0; ot < 4; ++ot) {
      v8f acc = {};
      acc = wmma16(av0, load_bfrag(wts[p], ot * 2 + 0, lane), acc);
      acc = wmma16(av1, load_bfrag(wts[p], ot * 2 + 1, lane), acc);
#pragma unroll
      for (int r = 0; r < 8; ++r) {
        int M = half * 8 + r, col = ot * 16 + ncol;
        outs[p][((size_t)(b * NN + row0 + M)) * DVZ + col] = acc[r];
      }
    }
  }
}

// ---------------- masked bus attention (online softmax, active keys only) ----------------
__global__ void k_attn(const float* __restrict__ q_g, const float* __restrict__ k_g,
                       const float* __restrict__ vh_g, const float* __restrict__ mask_g,
                       float* __restrict__ mraw) {
  int idx = blockIdx.x * blockDim.x + threadIdx.x;  // (b, head, query)
  if (idx >= BB * NH * NN) return;
  int qi = idx % NN;
  int hh = (idx / NN) % NH;
  int b  = idx / (NN * NH);
  const float* qv = q_g + ((size_t)(b * NN + qi)) * DVZ + hh * HDZ;
  float qr[HDZ];
#pragma unroll
  for (int d = 0; d < HDZ; ++d) qr[d] = qv[d];
  const float* mk = mask_g + (size_t)b * NN;
  float mx = -1e30f, sum = 0.f, o[HDZ];
#pragma unroll
  for (int d = 0; d < HDZ; ++d) o[d] = 0.f;
  for (int n = 0; n < NN; ++n) {
    if (mk[n] == 0.0f) continue;                 // -inf logit => excluded
    const float* kv = k_g + ((size_t)(b * NN + n)) * DVZ + hh * HDZ;
    float l = 0.f;
#pragma unroll
    for (int d = 0; d < HDZ; ++d) l += qr[d] * kv[d];
    l *= 0.25f;                                   // 1/sqrt(HD=16)
    float nm = fmaxf(mx, l);
    float sc = __expf(mx - nm);
    float w  = __expf(l - nm);
    sum = sum * sc + w;
    const float* vv = vh_g + ((size_t)(b * NN + n)) * DVZ + hh * HDZ;
#pragma unroll
    for (int d = 0; d < HDZ; ++d) o[d] = o[d] * sc + w * vv[d];
    mx = nm;
  }
  float inv = 1.0f / sum;                         // K=512 active keys => sum > 0
  float* mr = mraw + ((size_t)(b * NN + qi)) * DVZ + hh * HDZ;
#pragma unroll
  for (int d = 0; d < HDZ; ++d) mr[d] = o[d] * inv;
}

// ---------------- attn-out projection + FiLM (WMMA) ----------------
__global__ void __launch_bounds__(32)
k_film(float* __restrict__ h_g, const float* __restrict__ mraw,
       const _Float16* __restrict__ wao16, const _Float16* __restrict__ wfs16,
       const _Float16* __restrict__ wfsh16) {
  __shared__ float ms_[16 * 64];
  __shared__ float mm[16 * 64];
  int lane = threadIdx.x;
  int b    = blockIdx.x / (NN / 16);
  int tile = blockIdx.x % (NN / 16);
  int row0 = tile * 16;
  for (int i = lane; i < 16 * 64; i += 32) {
    int r = i >> 6, c = i & 63;
    ms_[i] = mraw[((size_t)(b * NN + row0 + r)) * DVZ + c];
  }
  __syncthreads();
  int ncol = lane & 15, half = lane >> 4;
  v16h am0 = load_afrag(ms_, 64, 0, lane);
  v16h am1 = load_afrag(ms_, 64, 1, lane);
#pragma unroll
  for (int ot = 0; ot < 4; ++ot) {                // m = mraw @ attn_out^T
    v8f acc = {};
    acc = wmma16(am0, load_bfrag(wao16, ot * 2 + 0, lane), acc);
    acc = wmma16(am1, load_bfrag(wao16, ot * 2 + 1, lane), acc);
#pragma unroll
    for (int r = 0; r < 8; ++r) mm[(half * 8 + r) * 64 + ot * 16 + ncol] = acc[r];
  }
  __syncthreads();
  v16h ao0 = load_afrag(mm, 64, 0, lane);
  v16h ao1 = load_afrag(mm, 64, 1, lane);
#pragma unroll
  for (int ot = 0; ot < 4; ++ot) {                // FiLM scale & shift
    v8f as_ = {}, ash = {};
    as_ = wmma16(ao0, load_bfrag(wfs16,  ot * 2 + 0, lane), as_);
    as_ = wmma16(ao1, load_bfrag(wfs16,  ot * 2 + 1, lane), as_);
    ash = wmma16(ao0, load_bfrag(wfsh16, ot * 2 + 0, lane), ash);
    ash = wmma16(ao1, load_bfrag(wfsh16, ot * 2 + 1, lane), ash);
#pragma unroll
    for (int r = 0; r < 8; ++r) {
      int M = half * 8 + r, col = ot * 16 + ncol;
      size_t gi = ((size_t)(b * NN + row0 + M)) * DSZ + col;
      float hv = h_g[gi];
      h_g[gi] = hv * (1.0f + tanhf(as_[r])) + ash[r];
    }
  }
}

// ---------------- learned-query pooling + output projection ----------------
__global__ void k_pool(const float* __restrict__ h_g, const float* __restrict__ query,
                       const float* __restrict__ out_norm_w, const float* __restrict__ W_out,
                       float* __restrict__ out, int t) {
  int b = blockIdx.x, tid = threadIdx.x;
  __shared__ float pl[NN];
  __shared__ float red[256];
  __shared__ float pooled[DSZ];
  __shared__ float qv[DSZ];
  if (tid < DSZ) qv[tid] = query[tid];
  __syncthreads();
  for (int n = tid; n < NN; n += 256) {
    const float* hr = h_g + ((size_t)(b * NN + n)) * DSZ;
    float s = 0.f;
#pragma unroll 8
    for (int d = 0; d < DSZ; ++d) s += qv[d] * hr[d];
    pl[n] = s * 0.125f;                           // 1/sqrt(DS=64)
  }
  __syncthreads();
  float m = -1e30f;
  for (int n = tid; n < NN; n += 256) m = fmaxf(m, pl[n]);
  red[tid] = m; __syncthreads();
  for (int s = 128; s > 0; s >>= 1) { if (tid < s) red[tid] = fmaxf(red[tid], red[tid + s]); __syncthreads(); }
  float mx = red[0]; __syncthreads();
  float ssum = 0.f;
  for (int n = tid; n < NN; n += 256) { float e = __expf(pl[n] - mx); pl[n] = e; ssum += e; }
  red[tid] = ssum; __syncthreads();
  for (int s = 128; s > 0; s >>= 1) { if (tid < s) red[tid] += red[tid + s]; __syncthreads(); }
  float inv = 1.0f / red[0]; __syncthreads();
  if (tid < DSZ) {
    float acc = 0.f;
    for (int n = 0; n < NN; ++n) acc += pl[n] * h_g[((size_t)(b * NN + n)) * DSZ + tid];
    pooled[tid] = acc * inv;
  }
  __syncthreads();
  if (tid == 0) {
    float ss = 0.f;
#pragma unroll 8
    for (int d = 0; d < DSZ; ++d) ss += pooled[d] * pooled[d];
    red[0] = rsqrtf(ss * (1.0f / DSZ) + 1e-6f);
  }
  __syncthreads();
  float rn = red[0];
  for (int o = tid; o < DIZ; o += 256) {
    float s = 0.f;
#pragma unroll 8
    for (int d = 0; d < DSZ; ++d)
      s += W_out[(size_t)o * DSZ + d] * (out_norm_w[d] * pooled[d] * rn);
    out[((size_t)b * TT + t) * DIZ + o] = s;
  }
}

// ---------------- host launch ----------------
extern "C" void kernel_launch(void* const* d_in, const int* in_sizes, int n_in,
                              void* d_out, int out_size, void* d_ws, size_t ws_size,
                              hipStream_t stream) {
  const float* x            = (const float*)d_in[0];
  const float* W_in         = (const float*)d_in[1];
  const float* core_norm_w  = (const float*)d_in[2];
  const float* core_Wg      = (const float*)d_in[3];
  const float* core_Wout    = (const float*)d_in[4];
  const float* W_gate       = (const float*)d_in[5];
  const float* W_vote       = (const float*)d_in[6];
  const float* attn_in_w    = (const float*)d_in[7];
  const float* attn_out_w   = (const float*)d_in[8];
  const float* W_film_scale = (const float*)d_in[9];
  const float* W_film_shift = (const float*)d_in[10];
  const float* query        = (const float*)d_in[11];
  const float* out_norm_w   = (const float*)d_in[12];
  const float* W_out        = (const float*)d_in[13];
  float* out = (float*)d_out;

  char* ws = (char*)d_ws;
  constexpr size_t SZ_BND = (size_t)BB * NN * DSZ * sizeof(float);  // 1 MB
  constexpr size_t oH    = 0;
  constexpr size_t oV    = oH  + SZ_BND;
  constexpr size_t oQ    = oV  + SZ_BND;
  constexpr size_t oK    = oQ  + SZ_BND;
  constexpr size_t oVH   = oK  + SZ_BND;
  constexpr size_t oM    = oVH + SZ_BND;
  constexpr size_t oGate = oM  + SZ_BND;
  constexpr size_t oMask = oGate + (size_t)BB * NN * sizeof(float);
  constexpr size_t oXp   = oMask + (size_t)BB * NN * sizeof(float);
  constexpr size_t oWG   = oXp + 1024;                       // 256B-aligned
  constexpr size_t oWOUT = oWG   + 128 * 512 * 2;
  constexpr size_t oWVO  = oWOUT + 256 * 64 * 2;
  constexpr size_t oWQ   = oWVO  + 64 * 64 * 2;
  constexpr size_t oWK   = oWQ   + 64 * 64 * 2;
  constexpr size_t oWV   = oWK   + 64 * 64 * 2;
  constexpr size_t oWAO  = oWV   + 64 * 64 * 2;
  constexpr size_t oWFS  = oWAO  + 64 * 64 * 2;
  constexpr size_t oWFSH = oWFS  + 64 * 64 * 2;

  float* h     = (float*)(ws + oH);
  float* v     = (float*)(ws + oV);
  float* q     = (float*)(ws + oQ);
  float* kk    = (float*)(ws + oK);
  float* vh    = (float*)(ws + oVH);
  float* mraw  = (float*)(ws + oM);
  float* gate  = (float*)(ws + oGate);
  float* mask  = (float*)(ws + oMask);
  float* xproj = (float*)(ws + oXp);
  _Float16* wg16   = (_Float16*)(ws + oWG);
  _Float16* wout16 = (_Float16*)(ws + oWOUT);
  _Float16* wvo16  = (_Float16*)(ws + oWVO);
  _Float16* wq16   = (_Float16*)(ws + oWQ);
  _Float16* wk16   = (_Float16*)(ws + oWK);
  _Float16* wv16   = (_Float16*)(ws + oWV);
  _Float16* wao16  = (_Float16*)(ws + oWAO);
  _Float16* wfs16  = (_Float16*)(ws + oWFS);
  _Float16* wfsh16 = (_Float16*)(ws + oWFSH);

  hipMemsetAsync(h, 0, SZ_BND, stream);  // h0 = zeros (reset_state per forward)

  // weight prep (B-fragment layout, f16)
  k_prep<<<32 * 4, 32, 0, stream>>>(core_Wg,      wg16,   4, 192, 0);
  k_prep<<<4 * 8,  32, 0, stream>>>(core_Wout,    wout16, 8, 256, 0);
  k_prep<<<4 * 2,  32, 0, stream>>>(W_vote,       wvo16,  2, 64,  0);
  k_prep<<<4 * 2,  32, 0, stream>>>(attn_in_w,    wq16,   2, 64,  0);
  k_prep<<<4 * 2,  32, 0, stream>>>(attn_in_w,    wk16,   2, 64,  64);
  k_prep<<<4 * 2,  32, 0, stream>>>(attn_in_w,    wv16,   2, 64,  128);
  k_prep<<<4 * 2,  32, 0, stream>>>(attn_out_w,   wao16,  2, 64,  0);
  k_prep<<<4 * 2,  32, 0, stream>>>(W_film_scale, wfs16,  2, 64,  0);
  k_prep<<<4 * 2,  32, 0, stream>>>(W_film_shift, wfsh16, 2, 64,  0);

  const int gateThreads = BB * NN + BB * DSZ;
  for (int t = 0; t < TT; ++t) {
    k_gate<<<(gateThreads + 255) / 256, 256, 0, stream>>>(x, W_in, W_gate, h, gate, xproj, t);
    k_topk<<<BB, 256, 0, stream>>>(gate, mask);
    k_core<<<BB * (NN / 16), 32, 0, stream>>>(h, xproj, mask, core_norm_w,
                                              wg16, wout16, wvo16, wq16, wk16, wv16,
                                              v, q, kk, vh);
    k_attn<<<(BB * NH * NN + 255) / 256, 256, 0, stream>>>(q, kk, vh, mask, mraw);
    k_film<<<BB * (NN / 16), 32, 0, stream>>>(h, mraw, wao16, wfs16, wfsh16);
    k_pool<<<BB, 256, 0, stream>>>(h, query, out_norm_w, W_out, out, t);
  }
}
